// GCN_25520695673511
// MI455X (gfx1250) — matrix operand order repaired and checked by reading, not compiled, not run on hardware
//
#include <hip/hip_runtime.h>

// ---------------------------------------------------------------------------
// GCN forward for MI455X (gfx1250, wave32).
//  * GEMMs: v_wmma_f32_16x16x32_f16; B staged in LDS, 8 fragments batched
//    per k-step to overlap ds_load latency with the WMMA pipe.
//  * Aggregation: CSR-by-destination (built once per call); one wave per node,
//    register accumulation, 4-edge unroll, no float atomics. Fuses
//    self-loop + bias + relu + f16 cast.
//  * Pooling: sorted-batch segment reduction (binary-searched bounds),
//    one wave per graph, no atomics, deterministic.
// ---------------------------------------------------------------------------

typedef __attribute__((ext_vector_type(16))) _Float16 v16h;
typedef __attribute__((ext_vector_type(8)))  _Float16 v8h;
typedef __attribute__((ext_vector_type(4)))  _Float16 v4h;
typedef __attribute__((ext_vector_type(8)))  float    v8f;

#define HDIM   128    // F == H == 128
#define SCAN_B 256

// ---------------- CSR build: histogram, scan, fill ----------------

__global__ void k_zero_int(int* __restrict__ p, int n) {
    int i = blockIdx.x * blockDim.x + threadIdx.x;
    if (i < n) p[i] = 0;
}

__global__ void k_count(const int* __restrict__ dst, int* __restrict__ cnt, int E) {
    int e = blockIdx.x * blockDim.x + threadIdx.x;
    if (e < E) atomicAdd(&cnt[dst[e]], 1);
}

__global__ __launch_bounds__(SCAN_B)
void k_block_sum(const int* __restrict__ cnt, int* __restrict__ bsum, int N) {
    __shared__ int sd[SCAN_B];
    int i = blockIdx.x * SCAN_B + threadIdx.x;
    sd[threadIdx.x] = (i < N) ? cnt[i] : 0;
    __syncthreads();
    for (int s = SCAN_B / 2; s > 0; s >>= 1) {
        if (threadIdx.x < s) sd[threadIdx.x] += sd[threadIdx.x + s];
        __syncthreads();
    }
    if (threadIdx.x == 0) bsum[blockIdx.x] = sd[0];
}

__global__ void k_scan_bsum(int* __restrict__ bsum, int nb) {
    if (blockIdx.x == 0 && threadIdx.x == 0) {
        int run = 0;
        for (int i = 0; i < nb; ++i) { int v = bsum[i]; bsum[i] = run; run += v; }
    }
}

__global__ __launch_bounds__(SCAN_B)
void k_block_scan(const int* __restrict__ cnt, const int* __restrict__ bsum,
                  int* __restrict__ rowstart, int* __restrict__ cursor, int N) {
    __shared__ int sd[SCAN_B];
    int i = blockIdx.x * SCAN_B + threadIdx.x;
    int v = (i < N) ? cnt[i] : 0;
    sd[threadIdx.x] = v;
    __syncthreads();
    for (int s = 1; s < SCAN_B; s <<= 1) {                 // inclusive Hillis-Steele
        int add = (threadIdx.x >= s) ? sd[threadIdx.x - s] : 0;
        __syncthreads();
        sd[threadIdx.x] += add;
        __syncthreads();
    }
    if (i < N) {
        int ex = sd[threadIdx.x] - v + bsum[blockIdx.x];   // exclusive
        rowstart[i] = ex;
        cursor[i]   = ex;
    }
}

__global__ void k_fill(const int* __restrict__ src, const int* __restrict__ dst,
                       int* __restrict__ cursor, int* __restrict__ csr_src, int E) {
    int e = blockIdx.x * blockDim.x + threadIdx.x;
    if (e < E) {
        int pos = atomicAdd(&cursor[dst[e]], 1);
        csr_src[pos] = src[e];
    }
}

// dinv = rsqrt(indegree + 1)   (self-loop included; always > 0)
__global__ void k_dinv(const int* __restrict__ cnt, float* __restrict__ dinv, int N) {
    int i = blockIdx.x * blockDim.x + threadIdx.x;
    if (i < N) dinv[i] = rsqrtf((float)cnt[i] + 1.0f);
}

// ---------------- casts ----------------

__global__ void k_cast_f16(const float* __restrict__ x, _Float16* __restrict__ o, int n) {
    int i = blockIdx.x * blockDim.x + threadIdx.x;
    if (i < n) o[i] = (_Float16)x[i];
}

// W [K=128][N=128] row-major f32  ->  Wt [N=128][K=128] col-major f16
__global__ void k_castW_t(const float* __restrict__ W, _Float16* __restrict__ Wt) {
    int idx = blockIdx.x * blockDim.x + threadIdx.x;
    if (idx < HDIM * HDIM) {
        int k = idx >> 7, n = idx & 127;
        Wt[n * HDIM + k] = (_Float16)W[k * HDIM + n];
    }
}

// ---------------- WMMA GEMM:  hw[N,128] = h16[N,128] @ W ----------------
// B (32KB f16, col-major) staged in LDS per block; one wave per 16x128 strip.
// All 8 B fragments of a k-step are loaded before the WMMA chain so the
// ds_load latency overlaps the matrix pipe. A/B use contiguous k (k = 16*hi + s) in both; C layout per CDNA5 ISA 7.12.2.
__global__ __launch_bounds__(256)
void k_gemm_wmma(const _Float16* __restrict__ A,      // [N][128] row-major
                 const _Float16* __restrict__ Bt,     // [128 n][128 k] col-major
                 float* __restrict__ C, int N) {
    __shared__ __attribute__((aligned(32))) _Float16 lB[HDIM * HDIM];   // 32 KB

    for (int o = threadIdx.x * 8; o < HDIM * HDIM; o += 256 * 8)
        *(v8h*)(lB + o) = *(const v8h*)(Bt + o);
    __syncthreads();

    int wave    = (blockIdx.x * blockDim.x + threadIdx.x) >> 5;
    int lane    = threadIdx.x & 31;
    int rowBase = wave * 16;
    if (rowBase < N) {
        int m  = lane & 15;
        int hi = lane >> 4;
        int arow = rowBase + m;
        if (arow >= N) arow = N - 1;                  // clamp (harmless dup read)
        const _Float16* ap = A + (size_t)arow * HDIM;

        v8f acc[8];
#pragma unroll
        for (int t = 0; t < 8; ++t) acc[t] = (v8f){0.f,0.f,0.f,0.f,0.f,0.f,0.f,0.f};

#pragma unroll
        for (int k = 0; k < HDIM; k += 32) {
            v8h a0 = *(const v8h*)(ap + k + hi * 16);
            v8h a1 = *(const v8h*)(ap + k + hi * 16 + 8);
            v16h afrag;
#pragma unroll
            for (int t = 0; t < 8; ++t) { afrag[t] = a0[t]; afrag[t + 8] = a1[t]; }

            v16h bf[8];
#pragma unroll
            for (int nt = 0; nt < 8; ++nt)
                bf[nt] = *(const v16h*)(lB + (nt * 16 + m) * HDIM + k + hi * 16);
#pragma unroll
            for (int nt = 0; nt < 8; ++nt)
                acc[nt] = __builtin_amdgcn_wmma_f32_16x16x32_f16(
                    false, afrag, false, bf[nt], (short)0, acc[nt], false, false);
        }

#pragma unroll
        for (int nt = 0; nt < 8; ++nt) {
            int n = nt * 16 + m;
#pragma unroll
            for (int v = 0; v < 8; ++v) {
                int r = rowBase + hi * 8 + v;
                if (r < N) C[(size_t)r * HDIM + n] = acc[nt][v];
            }
        }
    }
}

// ---------------- fused aggregation ----------------
// One wave per destination node: agg = dinv_i^2*hw[i] + sum_e dinv_s*dinv_i*hw[s]
// then h = relu(agg + b); writes f32 (pooling) and f16 (next GEMM). No atomics.
// 4-edge unroll keeps 4 independent 512B row gathers in flight.
__global__ __launch_bounds__(256)
void k_aggregate(const float* __restrict__ hw, const float* __restrict__ dinv,
                 const int* __restrict__ rowstart, const int* __restrict__ cnt,
                 const int* __restrict__ csr_src, const float* __restrict__ b,
                 float* __restrict__ hout, _Float16* __restrict__ h16, int N) {
    int i    = (blockIdx.x * blockDim.x + threadIdx.x) >> 5;
    int lane = threadIdx.x & 31;
    if (i >= N) return;

    float di = dinv[i];
    float4 v = *((const float4*)(hw + (size_t)i * HDIM) + lane);   // self row
    float w = di * di;
    float ax = v.x * w, ay = v.y * w, az = v.z * w, aw = v.w * w;

    int j   = rowstart[i];
    int end = j + cnt[i];
    for (; j + 4 <= end; j += 4) {
        int s0 = csr_src[j],     s1 = csr_src[j + 1];
        int s2 = csr_src[j + 2], s3 = csr_src[j + 3];
        float n0 = dinv[s0] * di, n1 = dinv[s1] * di;
        float n2 = dinv[s2] * di, n3 = dinv[s3] * di;
        float4 u0 = *((const float4*)(hw + (size_t)s0 * HDIM) + lane);
        float4 u1 = *((const float4*)(hw + (size_t)s1 * HDIM) + lane);
        float4 u2 = *((const float4*)(hw + (size_t)s2 * HDIM) + lane);
        float4 u3 = *((const float4*)(hw + (size_t)s3 * HDIM) + lane);
        ax += u0.x * n0 + u1.x * n1 + u2.x * n2 + u3.x * n3;
        ay += u0.y * n0 + u1.y * n1 + u2.y * n2 + u3.y * n3;
        az += u0.z * n0 + u1.z * n1 + u2.z * n2 + u3.z * n3;
        aw += u0.w * n0 + u1.w * n1 + u2.w * n2 + u3.w * n3;
    }
    for (; j < end; ++j) {
        int s = csr_src[j];
        float nrm = dinv[s] * di;
        float4 u = *((const float4*)(hw + (size_t)s * HDIM) + lane);
        ax += u.x * nrm; ay += u.y * nrm; az += u.z * nrm; aw += u.w * nrm;
    }

    float4 bb = *((const float4*)b + lane);
    ax = fmaxf(ax + bb.x, 0.0f);
    ay = fmaxf(ay + bb.y, 0.0f);
    az = fmaxf(az + bb.z, 0.0f);
    aw = fmaxf(aw + bb.w, 0.0f);

    *((float4*)(hout + (size_t)i * HDIM) + lane) = make_float4(ax, ay, az, aw);
    v4h hv; hv[0] = (_Float16)ax; hv[1] = (_Float16)ay; hv[2] = (_Float16)az; hv[3] = (_Float16)aw;
    *((v4h*)(h16 + (size_t)i * HDIM) + lane) = hv;
}

// ---------------- pooling (sorted batch -> segment bounds) ----------------

// gstart[g] = first node index with batch[i] >= g, for g = 0..G (gstart[G] = N)
__global__ void k_gbounds(const int* __restrict__ batch, int* __restrict__ gstart,
                          int N, int G) {
    int g = blockIdx.x * blockDim.x + threadIdx.x;
    if (g > G) return;
    int lo = 0, hi = N;
    while (lo < hi) {
        int mid = (lo + hi) >> 1;
        if (batch[mid] < g) lo = mid + 1; else hi = mid;
    }
    gstart[g] = lo;
}

// one wave per graph: sum contiguous node rows (coalesced, unrolled x4)
__global__ __launch_bounds__(256)
void k_pool(const float* __restrict__ h, const int* __restrict__ gstart,
            float* __restrict__ pooled, int G) {
    int g    = (blockIdx.x * blockDim.x + threadIdx.x) >> 5;
    int lane = threadIdx.x & 31;
    if (g >= G) return;
    int i   = gstart[g];
    int end = gstart[g + 1];
    float ax = 0.f, ay = 0.f, az = 0.f, aw = 0.f;
    for (; i + 4 <= end; i += 4) {
        float4 u0 = *((const float4*)(h + (size_t)(i    ) * HDIM) + lane);
        float4 u1 = *((const float4*)(h + (size_t)(i + 1) * HDIM) + lane);
        float4 u2 = *((const float4*)(h + (size_t)(i + 2) * HDIM) + lane);
        float4 u3 = *((const float4*)(h + (size_t)(i + 3) * HDIM) + lane);
        ax += u0.x + u1.x + u2.x + u3.x;
        ay += u0.y + u1.y + u2.y + u3.y;
        az += u0.z + u1.z + u2.z + u3.z;
        aw += u0.w + u1.w + u2.w + u3.w;
    }
    for (; i < end; ++i) {
        float4 u = *((const float4*)(h + (size_t)i * HDIM) + lane);
        ax += u.x; ay += u.y; az += u.z; aw += u.w;
    }
    *((float4*)(pooled + (size_t)g * HDIM) + lane) = make_float4(ax, ay, az, aw);
}

// ---------------- head: mean + linear + log_softmax ----------------

__global__ void k_head(const float* __restrict__ pooled, const int* __restrict__ gstart,
                       const float* __restrict__ Wo, const float* __restrict__ bo,
                       float* __restrict__ out, int C) {
    __shared__ float ph[HDIM];
    __shared__ float lg[16];
    __shared__ float red[2];
    int g = blockIdx.x, t = threadIdx.x;               // blockDim = 128
    float cntf = (float)(gstart[g + 1] - gstart[g]);
    float inv  = 1.0f / fmaxf(cntf, 1.0f);
    ph[t] = pooled[(size_t)g * HDIM + t] * inv;
    __syncthreads();
    if (t < C) {
        float acc = bo[t];
        for (int j = 0; j < HDIM; ++j) acc += ph[j] * Wo[j * C + t];
        lg[t] = acc;
    }
    __syncthreads();
    if (t == 0) {
        float m = -3.4e38f;
        for (int c = 0; c < C; ++c) m = fmaxf(m, lg[c]);
        float s = 0.0f;
        for (int c = 0; c < C; ++c) s += expf(lg[c] - m);
        red[0] = m; red[1] = logf(s);
    }
    __syncthreads();
    if (t < C) out[(size_t)g * C + t] = lg[t] - red[0] - red[1];
}

// ---------------- launcher ----------------

extern "C" void kernel_launch(void* const* d_in, const int* in_sizes, int n_in,
                              void* d_out, int out_size, void* d_ws, size_t ws_size,
                              hipStream_t stream) {
    const float* x     = (const float*)d_in[0];
    const int*   edges = (const int*)d_in[1];          // [2][E] flat: src then dst
    const int*   batch = (const int*)d_in[2];
    const float* W[3]  = { (const float*)d_in[4], (const float*)d_in[6], (const float*)d_in[8] };
    const float* b[3]  = { (const float*)d_in[5], (const float*)d_in[7], (const float*)d_in[9] };
    const float* Wo    = (const float*)d_in[10];
    const float* bo    = (const float*)d_in[11];
    float*       out   = (float*)d_out;

    const int N = in_sizes[0] / HDIM;
    const int E = in_sizes[1] / 2;
    const int C = in_sizes[11];
    const int G = out_size / C;
    const int* src = edges;
    const int* dst = edges + E;

    // carve workspace (256B aligned regions)
    char* ws = (char*)d_ws; size_t off = 0;
    auto carve = [&](size_t bytes) -> void* {
        void* p = ws + off;
        off += (bytes + 255) & ~(size_t)255;
        return p;
    };
    int*      cnt      = (int*)     carve((size_t)N * 4);
    int*      rowstart = (int*)     carve((size_t)N * 4);
    int*      cursor   = (int*)     carve((size_t)N * 4);
    int*      bsum     = (int*)     carve(((size_t)N / SCAN_B + 2) * 4);
    int*      gstart   = (int*)     carve((size_t)(G + 1) * 4);
    int*      csr_src  = (int*)     carve((size_t)E * 4);
    float*    dinv     = (float*)   carve((size_t)N * 4);
    float*    hw       = (float*)   carve((size_t)N * HDIM * 4);
    float*    hf       = (float*)   carve((size_t)N * HDIM * 4);
    _Float16* h16      = (_Float16*)carve((size_t)N * HDIM * 2);
    _Float16* Wt       = (_Float16*)carve((size_t)HDIM * HDIM * 2);
    float*    pooled   = (float*)   carve((size_t)G * HDIM * 4);
    (void)ws_size;

    const int TB = 256;
    const int NH = N * HDIM;
    const int nb = (N + SCAN_B - 1) / SCAN_B;

    // ---- CSR build + normalization + graph bounds (once per call) ----
    k_zero_int  <<<(N + TB - 1) / TB, TB, 0, stream>>>(cnt, N);
    k_count     <<<(E + TB - 1) / TB, TB, 0, stream>>>(dst, cnt, E);
    k_block_sum <<<nb, SCAN_B, 0, stream>>>(cnt, bsum, N);
    k_scan_bsum <<<1, 32, 0, stream>>>(bsum, nb);
    k_block_scan<<<nb, SCAN_B, 0, stream>>>(cnt, bsum, rowstart, cursor, N);
    k_dinv      <<<(N + TB - 1) / TB, TB, 0, stream>>>(cnt, dinv, N);
    k_fill      <<<(E + TB - 1) / TB, TB, 0, stream>>>(src, dst, cursor, csr_src, E);
    k_gbounds   <<<(G + 1 + TB - 1) / TB, TB, 0, stream>>>(batch, gstart, N, G);

    // ---- layer input ----
    k_cast_f16<<<(NH + TB - 1) / TB, TB, 0, stream>>>(x, h16, NH);

    const int tiles      = (N + 15) / 16;
    const int gemmBlocks = (tiles + 7) / 8;            // 8 waves / block
    const int aggBlocks  = (N + 7) / 8;                // 1 wave per node

    for (int l = 0; l < 3; ++l) {
        k_castW_t  <<<(HDIM * HDIM + TB - 1) / TB, TB, 0, stream>>>(W[l], Wt);
        k_gemm_wmma<<<gemmBlocks, TB, 0, stream>>>(h16, Wt, hw, N);
        k_aggregate<<<aggBlocks, TB, 0, stream>>>(hw, dinv, rowstart, cnt, csr_src,
                                                  b[l], hf, h16, N);
    }

    // ---- global mean pool + head ----
    k_pool<<<(G + 7) / 8, TB, 0, stream>>>(hf, gstart, pooled, G);
    k_head<<<G, HDIM, 0, stream>>>(pooled, gstart, Wo, bo, out, C);
}
